// MultiHeadAttention_60687887893051
// MI455X (gfx1250) — matrix operand order repaired and checked by reading, not compiled, not run on hardware
//
#include <hip/hip_runtime.h>
#include <hip/hip_bf16.h>

#define N_HEADS 16
#define D_MODEL 1024
#define D_K     64
#define BATCH   4
#define SEQ     2048
#define MROWS   (BATCH*SEQ)   // 8192
#define NEG_BIG (-1.0e10f)

// padded LDS row strides (elements) -> conflict-free b128 fragment reads
#define AB_STR  40    // 80 B rows for 128x32 GEMM tiles
#define KS_STR  72    // 144 B rows for 128x64 K tile
#define VS_STR  136   // 272 B rows for 64x128 V^T tile
#define PS_STR  136   // 272 B rows for 16x128 P tile

typedef __attribute__((ext_vector_type(16))) __bf16 v16bf;
typedef __attribute__((ext_vector_type(8)))  __bf16 v8bf;
typedef __attribute__((ext_vector_type(8)))  float  v8f;
typedef __attribute__((ext_vector_type(4)))  float  v4f;

static __device__ __forceinline__ v16bf join8(v8bf lo, v8bf hi) {
  v16bf r;
#pragma unroll
  for (int i = 0; i < 8; ++i) { r[i] = lo[i]; r[i + 8] = hi[i]; }
  return r;
}

static __device__ __forceinline__ v8f zero8() {
  v8f z;
#pragma unroll
  for (int i = 0; i < 8; ++i) z[i] = 0.0f;
  return z;
}

// CDNA5 async global->LDS copy (16B per lane), tracked by ASYNCcnt.
static __device__ __forceinline__ void async_b128(void* lds_dst, const void* gsrc) {
  asm volatile("global_load_async_to_lds_b128 %0, %1, off"
               :: "v"((unsigned)(size_t)lds_dst), "v"(gsrc)
               : "memory");
}
static __device__ __forceinline__ void wait_async0() {
  asm volatile("s_wait_asynccnt 0" ::: "memory");
}

// ---------------------------------------------------------------------------
// Streaming f32 -> bf16 conversion (one-time pre-pass for X inputs + weights)
// ---------------------------------------------------------------------------
__global__ __launch_bounds__(256)
void cvt_f32_to_bf16(const float* __restrict__ in, __bf16* __restrict__ out, int n)
{
  const int i = (blockIdx.x * 256 + threadIdx.x) * 8;
  if (i < n) {
    v4f a = *(const v4f*)(in + i);
    v4f b = *(const v4f*)(in + i + 4);
    v8bf o;
#pragma unroll
    for (int u = 0; u < 4; ++u) { o[u] = (__bf16)a[u]; o[4 + u] = (__bf16)b[u]; }
    *(v8bf*)(out + i) = o;
  }
}

// ---------------------------------------------------------------------------
// C = X[M,K] @ W[N,K]^T + bias, X/W bf16      (M=8192, N=K=1024)
// MODE 0: fp32 out[M, D_MODEL]                (final projection -> d_out)
// MODE 1: bf16 out[B, H, S, D_K]              (Q / K split-head layout)
// MODE 2: bf16 out[B, H, D_K, S]              (V stored transposed)
// 256 thr = 8 waves; block tile 128x128; wave owns a 16-row strip.
// A/B tiles staged by async DMA, double-buffered to overlap with WMMA.
// ---------------------------------------------------------------------------
template <int MODE>
__global__ __launch_bounds__(256)
void gemm_bf16(const __bf16* __restrict__ X, const __bf16* __restrict__ W,
               const float* __restrict__ bias, void* __restrict__ out)
{
  __shared__ __bf16 As[2][128 * AB_STR];   // [m][k], padded rows
  __shared__ __bf16 Bs[2][128 * AB_STR];   // [n][k], padded rows

  const int tid  = threadIdx.x;
  const int wave = tid >> 5;
  const int lane = tid & 31;
  const int half = lane >> 4;
  const int lr   = lane & 15;
  const int m0   = blockIdx.y * 128;
  const int n0   = blockIdx.x * 128;
  const int K    = D_MODEL;
  const int NT   = K / 32;

  v8f acc[8];
#pragma unroll
  for (int t = 0; t < 8; ++t) acc[t] = zero8();

  // async-stage one 128x32 bf16 tile of X and W (512 16B chunks each)
  auto stage = [&](int k0, int buf) {
#pragma unroll
    for (int i = 0; i < 2; ++i) {
      const int c = tid + i * 256;         // 0..511
      const int r = c >> 2, p = c & 3;     // row, 16B piece within 64B data
      async_b128((char*)As[buf] + r * (AB_STR * 2) + p * 16,
                 (const char*)X + (((size_t)(m0 + r)) * K + k0) * 2 + p * 16);
      async_b128((char*)Bs[buf] + r * (AB_STR * 2) + p * 16,
                 (const char*)W + (((size_t)(n0 + r)) * K + k0) * 2 + p * 16);
    }
  };

  stage(0, 0);
  for (int kt = 0; kt < NT; ++kt) {
    wait_async0();
    __syncthreads();
    if (kt + 1 < NT) stage((kt + 1) * 32, (kt + 1) & 1);

    const __bf16* Ab = As[kt & 1];
    const __bf16* Bb = Bs[kt & 1];

    // A fragment: lanes 0-15: K {0..7,16..23}; lanes 16-31: K {8..15,24..31}
    const int arow = (wave * 16 + lr) * AB_STR;
    v8bf alo = *(const v8bf*)&Ab[arow + half * 8];
    v8bf ahi = *(const v8bf*)&Ab[arow + 16 + half * 8];
    v16bf afrag = join8(alo, ahi);

#pragma unroll
    for (int t = 0; t < 8; ++t) {
      // B fragment: N = t*16 + lr; lanes 0-15: K {0..7,16..23}; lanes 16-31: K {8..15,24..31}
      const int boff = (t * 16 + lr) * AB_STR + half * 8;
      v8bf b0 = *(const v8bf*)&Bb[boff];
      v8bf b1 = *(const v8bf*)&Bb[boff + 16];
      v16bf bfrag = join8(b0, b1);
      acc[t] = __builtin_amdgcn_wmma_f32_16x16x32_bf16(
          false, afrag, false, bfrag, (short)0, acc[t], false, false);
    }
    __syncthreads();
  }

  // epilogue: C layout is VGPR i -> row i + 8*half, col = lane%16
#pragma unroll
  for (int t = 0; t < 8; ++t) {
    const int n = n0 + t * 16 + lr;
    const float bn = bias[n];
#pragma unroll
    for (int i = 0; i < 8; ++i) {
      const int m = m0 + wave * 16 + i + 8 * half;
      const float v = acc[t][i] + bn;
      if (MODE == 0) {
        ((float*)out)[(size_t)m * D_MODEL + n] = v;
      } else {
        const int b = m >> 11, s = m & (SEQ - 1);
        const int h = n >> 6,  d = n & (D_K - 1);
        if (MODE == 1)
          ((__bf16*)out)[(((size_t)(b * N_HEADS + h)) * SEQ + s) * D_K + d] = (__bf16)v;
        else
          ((__bf16*)out)[(((size_t)(b * N_HEADS + h)) * D_K + d) * SEQ + s] = (__bf16)v;
      }
    }
  }
}

// ---------------------------------------------------------------------------
// Flash attention per (b,h): softmax(Q K^T / 8, mask) @ V
// Qp, Kp: bf16 [B,H,S,64]; VpT: bf16 [B,H,64,S]; out bf16 [B,S,D_MODEL]
// 256 thr = 8 waves; block covers 128 q rows; wave owns 16 rows.
// K / V^T tiles: double-buffered async DMA overlapped with WMMA + softmax.
// ---------------------------------------------------------------------------
__global__ __launch_bounds__(256)
void flash_attn(const __bf16* __restrict__ Qp, const __bf16* __restrict__ Kp,
                const __bf16* __restrict__ VpT, const int* __restrict__ mask,
                __bf16* __restrict__ attn)
{
  __shared__ __bf16 Ks[2][128 * KS_STR];   // [key][d]  padded rows
  __shared__ __bf16 Vs[2][64 * VS_STR];    // [d][key]  padded rows
  __shared__ __bf16 Ps[8 * 16 * PS_STR];   // per-wave P tile, padded rows

  const int tid  = threadIdx.x;
  const int wave = tid >> 5;
  const int lane = tid & 31;
  const int half = lane >> 4;
  const int lr   = lane & 15;
  const int bh   = blockIdx.y;              // b*H + h
  const int b    = bh >> 4;
  const int h    = bh & 15;
  const int q0   = blockIdx.x * 128 + wave * 16;
  const float scale = 0.125f;               // 1/sqrt(D_K)
  const int NT = SEQ / 128;

  // Q A-fragments (2 chunks of K=32 across D_K=64) held in registers
  const __bf16* qrow = Qp + ((size_t)bh * SEQ + q0 + lr) * D_K;
  v16bf aQ[2];
#pragma unroll
  for (int c = 0; c < 2; ++c) {
    v8bf lo = *(const v8bf*)(qrow + 32 * c + half * 8);
    v8bf hi = *(const v8bf*)(qrow + 32 * c + 16 + half * 8);
    aQ[c] = join8(lo, hi);
  }

  float mrun[8], lrun[8];
  v8f Oacc[4];
#pragma unroll
  for (int i = 0; i < 8; ++i) { mrun[i] = -3.0e38f; lrun[i] = 0.0f; }
#pragma unroll
  for (int t = 0; t < 4; ++t) Oacc[t] = zero8();

  __bf16* pw = &Ps[wave * (16 * PS_STR)];

  auto stage_kv = [&](int k0, int buf) {
    const char* gK = (const char*)(Kp + ((size_t)bh * SEQ + k0) * D_K);
    char* lK = (char*)Ks[buf];
#pragma unroll
    for (int i = 0; i < 4; ++i) {          // 1024 chunks; K row = 8 x 16B
      const int c = tid + i * 256;
      const int r = c >> 3, p = c & 7;
      async_b128(lK + r * (KS_STR * 2) + p * 16, gK + c * 16);
    }
    char* lV = (char*)Vs[buf];
#pragma unroll
    for (int i = 0; i < 4; ++i) {          // 1024 chunks; V row = 16 x 16B
      const int c = tid + i * 256;
      const int d = c >> 4, p = c & 15;
      async_b128(lV + d * (VS_STR * 2) + p * 16,
                 (const char*)(VpT + ((size_t)bh * D_K + d) * SEQ + k0) + p * 16);
    }
  };

  stage_kv(0, 0);
  for (int kt = 0; kt < NT; ++kt) {
    const int k0 = kt * 128;
    wait_async0();
    __syncthreads();
    if (kt + 1 < NT) stage_kv((kt + 1) * 128, (kt + 1) & 1);

    const __bf16* Kb = Ks[kt & 1];
    const __bf16* Vb = Vs[kt & 1];

    // ---- S = Q K^T over this 128-key tile: 16 WMMAs -> 16x128 scores ----
    v8f Sacc[8];
#pragma unroll
    for (int t = 0; t < 8; ++t) Sacc[t] = zero8();
#pragma unroll
    for (int c = 0; c < 2; ++c) {
#pragma unroll
      for (int t = 0; t < 8; ++t) {
        const __bf16* krow = Kb + (t * 16 + lr) * KS_STR + 32 * c + half * 8;
        v8bf b0 = *(const v8bf*)(krow);
        v8bf b1 = *(const v8bf*)(krow + 16);
        v16bf bK = join8(b0, b1);
        Sacc[t] = __builtin_amdgcn_wmma_f32_16x16x32_bf16(
            false, aQ[c], false, bK, (short)0, Sacc[t], false, false);
      }
    }

    // ---- scale + mask + online softmax (row reduce across 16-lane halves) ----
#pragma unroll
    for (int i = 0; i < 8; ++i) {
      const int q = q0 + i + 8 * half;
      const size_t mrow = ((size_t)b * SEQ + q) * SEQ + k0 + lr;
      float sv[8];
      float mx = -3.0e38f;
#pragma unroll
      for (int t = 0; t < 8; ++t) {
        float s = Sacc[t][i] * scale;
        if (mask[mrow + t * 16] == 0) s = NEG_BIG;
        sv[t] = s;
        mx = fmaxf(mx, s);
      }
#pragma unroll
      for (int off = 8; off >= 1; off >>= 1)
        mx = fmaxf(mx, __shfl_xor(mx, off, 16));
      const float mnew  = fmaxf(mrun[i], mx);
      const float alpha = __expf(mrun[i] - mnew);
      float sum = 0.0f;
#pragma unroll
      for (int t = 0; t < 8; ++t) {
        const float p = __expf(sv[t] - mnew);
        sum += p;
        pw[(i + 8 * half) * PS_STR + t * 16 + lr] = (__bf16)p;
      }
#pragma unroll
      for (int off = 8; off >= 1; off >>= 1)
        sum += __shfl_xor(sum, off, 16);
      lrun[i] = lrun[i] * alpha + sum;
      mrun[i] = mnew;
#pragma unroll
      for (int t = 0; t < 4; ++t) Oacc[t][i] *= alpha;
    }
    __syncthreads();

    // ---- O += P @ V : P A-fragments and V B-fragments from LDS ----
#pragma unroll
    for (int kc = 0; kc < 4; ++kc) {
      const __bf16* prow = pw + lr * PS_STR + 32 * kc;
      v8bf lo = *(const v8bf*)(prow + half * 8);
      v8bf hi = *(const v8bf*)(prow + 16 + half * 8);
      v16bf aP = join8(lo, hi);
#pragma unroll
      for (int t = 0; t < 4; ++t) {
        const __bf16* vrow = Vb + (t * 16 + lr) * VS_STR + 32 * kc + half * 8;
        v8bf b0 = *(const v8bf*)(vrow);
        v8bf b1 = *(const v8bf*)(vrow + 16);
        v16bf bV = join8(b0, b1);
        Oacc[t] = __builtin_amdgcn_wmma_f32_16x16x32_bf16(
            false, aP, false, bV, (short)0, Oacc[t], false, false);
      }
    }
    __syncthreads();
  }

  // ---- finalize: O /= l, write heads-concatenated bf16 [B,S,D_MODEL] ----
#pragma unroll
  for (int t = 0; t < 4; ++t) {
    const int d = t * 16 + lr;
#pragma unroll
    for (int i = 0; i < 8; ++i) {
      const int q = q0 + i + 8 * half;
      attn[((size_t)b * SEQ + q) * D_MODEL + h * D_K + d] =
          (__bf16)(Oacc[t][i] / lrun[i]);
    }
  }
}

// ---------------------------------------------------------------------------
extern "C" void kernel_launch(void* const* d_in, const int* in_sizes, int n_in,
                              void* d_out, int out_size, void* d_ws, size_t ws_size,
                              hipStream_t stream)
{
  const float* query = (const float*)d_in[0];
  const float* key_  = (const float*)d_in[1];
  const float* value = (const float*)d_in[2];
  const int*   mask  = (const int*)d_in[3];
  const float* Wq = (const float*)d_in[4];
  const float* bq = (const float*)d_in[5];
  const float* Wk = (const float*)d_in[6];
  const float* bk = (const float*)d_in[7];
  const float* Wv = (const float*)d_in[8];
  const float* bv = (const float*)d_in[9];
  const float* Wo = (const float*)d_in[10];
  const float* bo = (const float*)d_in[11];

  char* ws = (char*)d_ws;
  const size_t MB = 1024ull * 1024ull;
  const size_t nX = (size_t)MROWS * D_MODEL;    // 8.39M elems
  const size_t nW = (size_t)D_MODEL * D_MODEL;  // 1.05M elems

  __bf16* Xq  = (__bf16*)(ws +   0 * MB);   // 16 MB each
  __bf16* Xk  = (__bf16*)(ws +  16 * MB);
  __bf16* Xv  = (__bf16*)(ws +  32 * MB);
  __bf16* Wqb = (__bf16*)(ws +  48 * MB);   // 2 MB each
  __bf16* Wkb = (__bf16*)(ws +  50 * MB);
  __bf16* Wvb = (__bf16*)(ws +  52 * MB);
  __bf16* Wob = (__bf16*)(ws +  54 * MB);
  __bf16* Qp  = (__bf16*)(ws +  56 * MB);   // 16 MB each
  __bf16* Kp  = (__bf16*)(ws +  72 * MB);
  __bf16* VpT = (__bf16*)(ws +  88 * MB);
  __bf16* attnB = (__bf16*)(ws + 104 * MB); // 16 MB

  dim3 blk(256);
  const int gX = (int)(nX / 2048), gW = (int)(nW / 2048);
  cvt_f32_to_bf16<<<gX, blk, 0, stream>>>(query, Xq, (int)nX);
  cvt_f32_to_bf16<<<gX, blk, 0, stream>>>(key_,  Xk, (int)nX);
  cvt_f32_to_bf16<<<gX, blk, 0, stream>>>(value, Xv, (int)nX);
  cvt_f32_to_bf16<<<gW, blk, 0, stream>>>(Wq, Wqb, (int)nW);
  cvt_f32_to_bf16<<<gW, blk, 0, stream>>>(Wk, Wkb, (int)nW);
  cvt_f32_to_bf16<<<gW, blk, 0, stream>>>(Wv, Wvb, (int)nW);
  cvt_f32_to_bf16<<<gW, blk, 0, stream>>>(Wo, Wob, (int)nW);

  dim3 ggrid(D_MODEL / 128, MROWS / 128);   // (8, 64)
  gemm_bf16<1><<<ggrid, blk, 0, stream>>>(Xq, Wqb, bq, (void*)Qp);
  gemm_bf16<1><<<ggrid, blk, 0, stream>>>(Xk, Wkb, bk, (void*)Kp);
  gemm_bf16<2><<<ggrid, blk, 0, stream>>>(Xv, Wvb, bv, (void*)VpT);

  dim3 agrid(SEQ / 128, BATCH * N_HEADS);   // (16, 64)
  flash_attn<<<agrid, blk, 0, stream>>>(Qp, Kp, VpT, mask, attnB);

  gemm_bf16<0><<<ggrid, blk, 0, stream>>>(attnB, Wob, bo, d_out);
}
